// S2VT_23562190586245
// MI455X (gfx1250) — matrix operand order repaired and checked by reading, not compiled, not run on hardware
//
#include <hip/hip_runtime.h>
#include <hip/hip_bf16.h>
#include <math.h>

// ---------------------------------------------------------------------------
// S2VT LSTM captioner forward for gfx1250 (MI455X).
// All GEMMs via v_wmma_f32_16x16x32_bf16 with pre-packed bf16 fragments for
// both A (activations) and B (weights): inner loop = loads + WMMA only.
// ---------------------------------------------------------------------------

typedef __attribute__((ext_vector_type(16))) __bf16 bf16x16;
typedef __attribute__((ext_vector_type(8)))  float  f32x8;

#define WPB 8          // waves per block (256 threads, wave32)
#define NT  4          // 16-col N tiles per wave (4x A reuse)

__device__ __forceinline__ unsigned short f2bf(float x) {
  unsigned u = __float_as_uint(x);
  u += 0x7FFFu + ((u >> 16) & 1u);          // round-to-nearest-even
  return (unsigned short)(u >> 16);
}

__device__ __forceinline__ float sigf(float x) {
  return 1.f / (1.f + __expf(-x));
}

// A-fragment k_local for packed element v, lane-half hi:
//   lanes 0-15 (hi=0): K runs [0..7] and [16..23]; lanes 16-31 (hi=1): +8
__device__ __forceinline__ int a_klocal(int v, int hi) {
  return (v < 8) ? (v + 8 * hi) : (v + 8 + 8 * hi);
}

// ---------------------------------------------------------------------------
// Pack fp32 weight W[Nsrc][K] into bf16 B-fragment order, zero-padded to Npack:
//   Bp[((nt*Kt + kb)*32 + lane)*16 + v] = bf16(W[nt*16+(lane&15)][kb*32+klocal])
// ---------------------------------------------------------------------------
__global__ __launch_bounds__(256) void pack_b(const float* __restrict__ W,
                                              unsigned short* __restrict__ Bp,
                                              int Nsrc, int Npack, int K) {
  size_t idx = (size_t)blockIdx.x * blockDim.x + threadIdx.x;
  size_t total = (size_t)Npack * K;
  if (idx >= total) return;
  int v    = (int)(idx & 15);
  size_t r = idx >> 4;
  int lane = (int)(r & 31); r >>= 5;
  int Kt   = K >> 5;
  int kb   = (int)(r % Kt);
  int nt   = (int)(r / Kt);
  int k = kb * 32 + a_klocal(v, lane >> 4);
  int n = nt * 16 + (lane & 15);
  Bp[idx] = (n < Nsrc) ? f2bf(W[(size_t)n * K + k]) : (unsigned short)0;
}

// ---------------------------------------------------------------------------
// Pack fp32 activations A[M][K] into bf16 A-fragment order:
//   Ap[((mt*Kt + kb)*32 + lane)*16 + v] = bf16(A[mt*16 + (lane&15)][kb*32 + klocal])
// ---------------------------------------------------------------------------
__global__ __launch_bounds__(256) void pack_a(const float* __restrict__ A,
                                              unsigned short* __restrict__ Ap,
                                              int M, int K) {
  size_t idx = (size_t)blockIdx.x * blockDim.x + threadIdx.x;
  size_t total = (size_t)M * K;
  if (idx >= total) return;
  int v    = (int)(idx & 15);
  size_t r = idx >> 4;
  int lane = (int)(r & 31); r >>= 5;
  int Kt   = K >> 5;
  int kb   = (int)(r % Kt);
  int mt   = (int)(r / Kt);
  int row  = mt * 16 + (lane & 15);
  int k    = kb * 32 + a_klocal(v, lane >> 4);
  Ap[idx] = f2bf(A[(size_t)row * K + k]);
}

// ---------------------------------------------------------------------------
// WMMA GEMM:  C[M][N] = Ap @ Bp (+bias)(+addsrc), bf16 fragments, f32 accum.
// One wave computes a 16 x (NT*16) tile. Inner loop: 1 A load, NT B loads,
// NT v_wmma_f32_16x16x32_bf16 per 32-K block.
// ---------------------------------------------------------------------------
__global__ __launch_bounds__(256) void wmma_gemm(const unsigned short* __restrict__ Ap,
                                                 const unsigned short* __restrict__ Bp,
                                                 const float* __restrict__ bias,
                                                 const float* __restrict__ addsrc,
                                                 float* __restrict__ C,
                                                 int M, int N, int K) {
  const int lane = threadIdx.x & 31;
  const int wave = (int)((blockIdx.x * blockDim.x + threadIdx.x) >> 5);
  const int Kt = K >> 5;
  const int mtiles = M >> 4;
  const int groups = (N >> 4) / NT;
  if (wave >= mtiles * groups) return;     // wave-uniform: EXEC stays all-ones
  const int mt = wave / groups;
  const int ng = wave - mt * groups;
  const int m0 = mt << 4;
  const int n0 = ng * (NT * 16);
  const int r  = lane & 15;                // col within N-tile for C/D layout
  const int hi = lane >> 4;

  f32x8 acc[NT];
  if (addsrc) {
#pragma unroll
    for (int j = 0; j < NT; ++j)
#pragma unroll
      for (int g = 0; g < 8; ++g)
        acc[j][g] = addsrc[(size_t)(m0 + g + 8 * hi) * N + n0 + j * 16 + r];
  } else {
#pragma unroll
    for (int j = 0; j < NT; ++j)
#pragma unroll
      for (int g = 0; g < 8; ++g) acc[j][g] = 0.f;
  }

  const unsigned short* abase = Ap + (size_t)mt * Kt * 512 + (size_t)lane * 16;
  const unsigned short* bbase = Bp + (size_t)(n0 >> 4) * Kt * 512 + (size_t)lane * 16;

  for (int kb = 0; kb < Kt; ++kb) {
    bf16x16 av = *(const bf16x16*)(abase + (size_t)kb * 512);
    __builtin_prefetch((const void*)(abase + (size_t)(kb + 16) * 512), 0, 1);
#pragma unroll
    for (int j = 0; j < NT; ++j) {
      const unsigned short* bj = bbase + ((size_t)j * Kt + kb) * 512;
      __builtin_prefetch((const void*)(bj + 16 * 512), 0, 1);   // global_prefetch_b8
      bf16x16 bv = *(const bf16x16*)bj;
      acc[j] = __builtin_amdgcn_wmma_f32_16x16x32_bf16(
          /*neg_a=*/false, av, /*neg_b=*/false, bv,
          /*c_mod=*/(short)0, acc[j], /*reuse_a=*/false, /*reuse_b=*/false);
    }
  }

#pragma unroll
  for (int j = 0; j < NT; ++j) {
    float bv = bias ? bias[n0 + j * 16 + r] : 0.f;
#pragma unroll
    for (int g = 0; g < 8; ++g)
      C[(size_t)(m0 + g + 8 * hi) * N + n0 + j * 16 + r] = acc[j][g] + bv;
  }
}

// ---------------------------------------------------------------------------
// Frame mean directly into packed A fragments: xbarP (M=64, K=2048)
// ---------------------------------------------------------------------------
__global__ __launch_bounds__(256) void frame_mean_packed(const float* __restrict__ img,
                                                         unsigned short* __restrict__ xbarP) {
  size_t idx = (size_t)blockIdx.x * blockDim.x + threadIdx.x;   // 64*2048
  if (idx >= (size_t)64 * 2048) return;
  int v    = (int)(idx & 15);
  size_t rr = idx >> 4;
  int lane = (int)(rr & 31); rr >>= 5;
  const int Kt = 2048 >> 5;
  int kb = (int)(rr % Kt);
  int mt = (int)(rr / Kt);
  int b = mt * 16 + (lane & 15);
  int j = kb * 32 + a_klocal(v, lane >> 4);
  const float* p = img + (size_t)b * 32 * 2048 + j;
  float s = 0.f;
  for (int n = 0; n < 32; ++n) s += p[(size_t)n * 2048];
  xbarP[idx] = f2bf(s * (1.f / 32.f));
}

// ---------------------------------------------------------------------------
// X0P packed fragments for [wv_t, feat], M = 27*64 rows, K = 2048
// ---------------------------------------------------------------------------
__global__ __launch_bounds__(256) void build_x0_packed(const int* __restrict__ cap,
                                                       const float* __restrict__ emb,
                                                       const float* __restrict__ feat,
                                                       unsigned short* __restrict__ X0P) {
  size_t idx = (size_t)blockIdx.x * blockDim.x + threadIdx.x;   // 1728*2048
  if (idx >= (size_t)1728 * 2048) return;
  int v    = (int)(idx & 15);
  size_t rr = idx >> 4;
  int lane = (int)(rr & 31); rr >>= 5;
  const int Kt = 2048 >> 5;
  int kb = (int)(rr % Kt);
  int mt = (int)(rr / Kt);
  int m = mt * 16 + (lane & 15);           // m = t*64 + b
  int k = kb * 32 + a_klocal(v, lane >> 4);
  int t = m >> 6, b = m & 63;
  float val;
  if (k < 1024) {
    int tok = cap[b * 28 + t];
    val = emb[(size_t)tok * 1024 + k];
  } else {
    val = feat[(size_t)b * 1024 + (k - 1024)];
  }
  X0P[idx] = f2bf(val);
}

__global__ __launch_bounds__(256) void bias_sum(const float* __restrict__ a,
                                                const float* __restrict__ b,
                                                float* __restrict__ o, int n) {
  int i = blockIdx.x * blockDim.x + threadIdx.x;
  if (i < n) o[i] = a[i] + b[i];
}

__global__ __launch_bounds__(256) void pad_copy(const float* __restrict__ in,
                                                float* __restrict__ out, int n, int npad) {
  int i = blockIdx.x * blockDim.x + threadIdx.x;
  if (i < npad) out[i] = (i < n) ? in[i] : 0.f;
}

// ---------------------------------------------------------------------------
// LSTM pointwise: gates [B][4H] (i|f|g|o); c fp32 in place; h written directly
// in packed bf16 A-fragment layout (h is only ever consumed by GEMMs).
// ---------------------------------------------------------------------------
__global__ __launch_bounds__(256) void lstm_cell(const float* __restrict__ G,
                                                 float* __restrict__ c,
                                                 unsigned short* __restrict__ hP,
                                                 int B, int H) {
  int idx = blockIdx.x * blockDim.x + threadIdx.x;
  if (idx >= B * H) return;
  int b = idx / H, j = idx - b * H;
  const float* g = G + (size_t)b * 4 * H;
  float iv = sigf(g[j]);
  float fv = sigf(g[H + j]);
  float gv = tanhf(g[2 * H + j]);
  float ov = sigf(g[3 * H + j]);
  float cn = fv * c[idx] + iv * gv;
  c[idx] = cn;
  float hn = ov * tanhf(cn);
  // packed A-fragment position for (row=b, k=j)
  int Kt = H >> 5;
  int mt = b >> 4, rr = b & 15;
  int kb = j >> 5, kl = j & 31;
  int hi = (kl >> 3) & 1;
  int v  = (kl >= 16) ? (kl - 16 - 8 * hi + 8) : (kl - 8 * hi);
  hP[(((size_t)mt * Kt + kb) * 32 + (rr + 16 * hi)) * 16 + v] = f2bf(hn);
}

// ---------------------------------------------------------------------------
// Row-wise log_softmax over first V columns of L (row stride Npad),
// write to out[b][t][V]
// ---------------------------------------------------------------------------
__global__ __launch_bounds__(256) void log_softmax_rows(const float* __restrict__ L,
                                                        float* __restrict__ out,
                                                        int t, int V, int Npad) {
  int b = blockIdx.x;
  int tid = threadIdx.x;
  const float* row = L + (size_t)b * Npad;
  __shared__ float red[256];

  float mx = -3.4e38f;
  for (int v = tid; v < V; v += 256) mx = fmaxf(mx, row[v]);
  red[tid] = mx; __syncthreads();
  for (int s = 128; s > 0; s >>= 1) {
    if (tid < s) red[tid] = fmaxf(red[tid], red[tid + s]);
    __syncthreads();
  }
  mx = red[0]; __syncthreads();

  float sum = 0.f;
  for (int v = tid; v < V; v += 256) sum += __expf(row[v] - mx);
  red[tid] = sum; __syncthreads();
  for (int s = 128; s > 0; s >>= 1) {
    if (tid < s) red[tid] += red[tid + s];
    __syncthreads();
  }
  float lse = mx + __logf(red[0]);

  float* orow = out + ((size_t)b * 27 + t) * V;
  for (int v = tid; v < V; v += 256) orow[v] = row[v] - lse;
}

// ---------------------------------------------------------------------------
// Host side
// ---------------------------------------------------------------------------
static inline int cdiv(long long a, long long b) { return (int)((a + b - 1) / b); }

extern "C" void kernel_launch(void* const* d_in, const int* in_sizes, int n_in,
                              void* d_out, int out_size, void* d_ws, size_t ws_size,
                              hipStream_t stream) {
  (void)in_sizes; (void)n_in; (void)out_size; (void)ws_size;

  const int B = 64, H = 1024, VID = 2048, V = 20000, T = 27;
  const int VP = 20032;                    // V padded so (VP/16) % NT == 0

  const float* img  = (const float*)d_in[0];
  const int*   cap  = (const int*)d_in[1];
  const float* emb  = (const float*)d_in[2];
  const float* Wio  = (const float*)d_in[3];
  const float* bio  = (const float*)d_in[4];
  const float* Wis  = (const float*)d_in[5];
  const float* bis  = (const float*)d_in[6];
  const float* Wimg = (const float*)d_in[7];
  const float* bimg = (const float*)d_in[8];
  const float* Wv2w = (const float*)d_in[9];
  const float* bv2w = (const float*)d_in[10];
  const float* Wih  = (const float*)d_in[11];
  const float* bih  = (const float*)d_in[12];
  const float* Whh  = (const float*)d_in[13];
  const float* bhh  = (const float*)d_in[14];
  float* out = (float*)d_out;

  char* p = (char*)d_ws;
  auto carve = [&](size_t bytes) -> void* {
    void* q = (void*)p;
    p += (bytes + 255) & ~(size_t)255;
    return q;
  };

  unsigned short* WihP  = (unsigned short*)carve((size_t)4 * H * 2 * H * 2);
  unsigned short* WhhP  = (unsigned short*)carve((size_t)4 * H * H * 2);
  unsigned short* WvP   = (unsigned short*)carve((size_t)VP * H * 2);
  unsigned short* WimgP = (unsigned short*)carve((size_t)H * VID * 2);
  unsigned short* WioP  = (unsigned short*)carve((size_t)H * H * 2);
  unsigned short* WisP  = (unsigned short*)carve((size_t)H * H * 2);
  float* bg     = (float*)carve((size_t)4 * H * 4);
  float* bvP    = (float*)carve((size_t)VP * 4);
  unsigned short* xbarP = (unsigned short*)carve((size_t)B * VID * 2);
  float* feat   = (float*)carve((size_t)B * H * 4);
  unsigned short* featP = (unsigned short*)carve((size_t)B * H * 2);
  float* hbuf   = (float*)carve((size_t)B * H * 4);
  float* cbuf   = (float*)carve((size_t)B * H * 4);
  unsigned short* hP    = (unsigned short*)carve((size_t)B * H * 2);
  unsigned short* X0P   = (unsigned short*)carve((size_t)T * B * VID * 2);
  float* G0     = (float*)carve((size_t)T * B * 4 * H * 4);
  float* Gs     = (float*)carve((size_t)B * 4 * H * 4);
  float* logits = (float*)carve((size_t)B * VP * 4);

  auto launch_pack_b = [&](const float* W, unsigned short* Bp, int Nsrc, int Npack, int K) {
    long long total = (long long)Npack * K;
    pack_b<<<cdiv(total, 256), 256, 0, stream>>>(W, Bp, Nsrc, Npack, K);
  };
  auto launch_pack_a = [&](const float* A, unsigned short* Ap, int M, int K) {
    long long total = (long long)M * K;
    pack_a<<<cdiv(total, 256), 256, 0, stream>>>(A, Ap, M, K);
  };
  auto launch_gemm = [&](const unsigned short* Ap, const unsigned short* Bp,
                         const float* bias, const float* addsrc, float* C,
                         int M, int N, int K) {
    int waves = (M >> 4) * ((N >> 4) / NT);
    wmma_gemm<<<cdiv(waves, WPB), WPB * 32, 0, stream>>>(Ap, Bp, bias, addsrc, C, M, N, K);
  };

  // Weight packing (idempotent; captured once in the graph).
  launch_pack_b(Wih,  WihP,  4 * H, 4 * H, 2 * H);
  launch_pack_b(Whh,  WhhP,  4 * H, 4 * H, H);
  launch_pack_b(Wv2w, WvP,   V,     VP,    H);
  launch_pack_b(Wimg, WimgP, H,     H,     VID);
  launch_pack_b(Wio,  WioP,  H,     H,     H);
  launch_pack_b(Wis,  WisP,  H,     H,     H);
  bias_sum<<<cdiv(4 * H, 256), 256, 0, stream>>>(bih, bhh, bg, 4 * H);
  pad_copy<<<cdiv(VP, 256), 256, 0, stream>>>(bv2w, bvP, V, VP);

  // feat = mean_frames(image) @ W_img^T + b_img ; h0/c0 from feat
  frame_mean_packed<<<cdiv((long long)B * VID, 256), 256, 0, stream>>>(img, xbarP);
  launch_gemm(xbarP, WimgP, bimg, nullptr, feat, B, H, VID);
  launch_pack_a(feat, featP, B, H);
  launch_gemm(featP, WioP, bio, nullptr, hbuf, B, H, H);
  launch_gemm(featP, WisP, bis, nullptr, cbuf, B, H, H);
  launch_pack_a(hbuf, hP, B, H);

  // Hoist input-dependent gate contributions for all 27 steps (parallel GEMM):
  // G0[t] = [wv_t, feat] @ W_ih^T + (b_ih + b_hh)
  build_x0_packed<<<cdiv((long long)T * B * VID, 256), 256, 0, stream>>>(cap, emb, feat, X0P);
  launch_gemm(X0P, WihP, bg, nullptr, G0, T * B, 4 * H, 2 * H);

  // Sequential recurrence: only h @ W_hh^T remains inside the loop.
  for (int t = 0; t < T; ++t) {
    launch_gemm(hP, WhhP, nullptr, G0 + (size_t)t * B * 4 * H, Gs, B, 4 * H, H);
    lstm_cell<<<cdiv((long long)B * H, 256), 256, 0, stream>>>(Gs, cbuf, hP, B, H);
    launch_gemm(hP, WvP, bvP, nullptr, logits, B, V, H);
    log_softmax_rows<<<B, 256, 0, stream>>>(logits, out, t, V, VP);
  }
}
